// CRF_54683523613018
// MI455X (gfx1250) — matrix operand order repaired and checked by reading, not compiled, not run on hardware
//
#include <hip/hip_runtime.h>

typedef __attribute__((ext_vector_type(16))) _Float16 v16h;
typedef __attribute__((ext_vector_type(8)))  _Float16 v8h;
typedef __attribute__((ext_vector_type(8)))  float    v8f;
typedef __attribute__((ext_vector_type(4)))  int      i4v;

#define NTAGS   64
#define SEQ     512
#define NBATCH  512
#define TILE_B  16
#define PAD_ID   0
#define START_ID 1
#define END_ID   2

// ---- CDNA5 async global->LDS support (guarded; falls back to reg staging) ----
#if __has_builtin(__builtin_amdgcn_global_load_async_to_lds_b128)
#define HAVE_ASYNC_LDS 1
typedef __attribute__((address_space(1))) i4v* gptr_b128;
typedef __attribute__((address_space(3))) i4v* lptr_b128;
__device__ __forceinline__ void async_load16(const void* g, void* l) {
  // generic->AS1/AS3 via integer casts: LDS generic VA truncates to 32-bit LDS
  // offset (flat-aperture rule: the LDS address is the low 32 bits).
  __builtin_amdgcn_global_load_async_to_lds_b128(
      (gptr_b128)(uintptr_t)g,
      (lptr_b128)(unsigned int)(uintptr_t)l,
      0, 0);
}
__device__ __forceinline__ void wait_async_le2() {
#if __has_builtin(__builtin_amdgcn_s_wait_asynccnt)
  __builtin_amdgcn_s_wait_asynccnt(2);
#else
  asm volatile("s_wait_asynccnt 0x2" ::: "memory");
#endif
}
__device__ __forceinline__ void wait_async_le0() {
#if __has_builtin(__builtin_amdgcn_s_wait_asynccnt)
  __builtin_amdgcn_s_wait_asynccnt(0);
#else
  asm volatile("s_wait_asynccnt 0x0" ::: "memory");
#endif
}
#else
#define HAVE_ASYNC_LDS 0
#endif

// ---------------------------------------------------------------------------
// Forward algorithm (partition function) via WMMA in the exp-domain.
// One block = 16 batch rows, 4 wave32s; wave w computes output tags [16w,16w+16).
// alpha kept in LDS; exp(T) B-fragments resident in registers across all steps.
// Emissions stream HBM -> LDS 4-deep ring (async direct-to-LDS when available).
// ---------------------------------------------------------------------------
__global__ __launch_bounds__(128) void crf_forward_wmma(
    const float* __restrict__ emis,
    const float* __restrict__ mask,
    const float* __restrict__ trans,
    float* __restrict__ out)
{
  __shared__ float    alphaS[TILE_B][NTAGS];   // current alpha (f32)
  __shared__ _Float16 expaS[TILE_B][NTAGS];    // exp(alpha - rowmax) (f16)
  __shared__ float    emS[4][TILE_B][NTAGS];   // emissions ring buffer
  __shared__ float    pmaxS[TILE_B][8];
  __shared__ float    srowS[TILE_B];
  __shared__ float    mvalS[TILE_B];
  __shared__ float    totS[TILE_B];

  const int tid  = threadIdx.x;
  const int lane = tid & 31;
  const int wv   = tid >> 5;               // wave id 0..3
  const int b0   = blockIdx.x * TILE_B;

  const int m  = tid >> 3;                 // staging row 0..15
  const int j0 = (tid & 7) * 8;            // staging col 0,8,...,56
  const float* emrow = emis + (size_t)(b0 + m) * SEQ * NTAGS + j0;

  // ---- B fragments: exp(transition) is exactly {0,1}; constant over time ----
  // 16-bit B 32x16 layout: lane L -> N = L&15; element h -> K = (h>>3)*16 + (L>=16?8:0) + (h&7)
  const int jc   = wv * 16 + (lane & 15);
  const int kofB = (lane >> 4) * 8;
  v16h bf0, bf1;
  #pragma unroll
  for (int e = 0; e < 16; ++e) {
    bf0[e] = (_Float16)__expf(trans[((e >> 3) * 16 + kofB + (e & 7))      * NTAGS + jc]); // K 0..31
    bf1[e] = (_Float16)__expf(trans[((e >> 3) * 16 + kofB + (e & 7) + 32) * NTAGS + jc]); // K 32..63
  }

  // ---- alpha(t=0) = T[START][j] + em[b,0,j] ----
  #pragma unroll
  for (int q = 0; q < 8; ++q)
    alphaS[m][j0 + q] = trans[START_ID * NTAGS + j0 + q] + emrow[q];

#if HAVE_ASYNC_LDS
  // ---- prologue: async-stage em(t=1,2) into ring ----
  #pragma unroll
  for (int tt = 1; tt <= 2; ++tt) {
    async_load16(emrow + (size_t)tt * NTAGS,     &emS[tt & 3][m][j0]);
    async_load16(emrow + (size_t)tt * NTAGS + 4, &emS[tt & 3][m][j0 + 4]);
  }
  wait_async_le2();            // t=1 complete; t=2 may remain in flight
  __syncthreads();
#else
  // ---- prologue: stage em(t=1,2) into ring, load em(t=3) into regs ----
  #pragma unroll
  for (int tt = 1; tt <= 2; ++tt) {
    const float4* ep = (const float4*)(emrow + (size_t)tt * NTAGS);
    *(float4*)&emS[tt & 3][m][j0]     = ep[0];
    *(float4*)&emS[tt & 3][m][j0 + 4] = ep[1];
  }
  float4 p0, p1;
  { const float4* ep = (const float4*)(emrow + (size_t)3 * NTAGS);
    p0 = ep[0]; p1 = ep[1]; }
  __syncthreads();
#endif

  for (int t = 1; t < SEQ; ++t) {
#if HAVE_ASYNC_LDS
    // issue async loads for em(t+2): ring distance 2, ASYNCcnt-disciplined
    if (t + 2 < SEQ) {
      async_load16(emrow + (size_t)(t + 2) * NTAGS,     &emS[(t + 2) & 3][m][j0]);
      async_load16(emrow + (size_t)(t + 2) * NTAGS + 4, &emS[(t + 2) & 3][m][j0 + 4]);
    }
#else
    // stage em(t+2) (loaded last iteration) into ring; issue loads for em(t+3)
    if (t + 2 < SEQ) {
      *(float4*)&emS[(t + 2) & 3][m][j0]     = p0;
      *(float4*)&emS[(t + 2) & 3][m][j0 + 4] = p1;
    }
    if (t + 3 < SEQ) {
      const float4* ep = (const float4*)(emrow + (size_t)(t + 3) * NTAGS);
      p0 = ep[0]; p1 = ep[1];
    }
#endif
    if (tid < TILE_B) mvalS[tid] = mask[(size_t)(b0 + tid) * SEQ + t];

    // ---- row max (2-level reduce) ----
    float4 a0 = *(const float4*)&alphaS[m][j0];
    float4 a1 = *(const float4*)&alphaS[m][j0 + 4];
    float pm = fmaxf(fmaxf(fmaxf(a0.x, a0.y), fmaxf(a0.z, a0.w)),
                     fmaxf(fmaxf(a1.x, a1.y), fmaxf(a1.z, a1.w)));
    pmaxS[m][tid & 7] = pm;
    __syncthreads();
    if (tid < TILE_B) {
      float s = pmaxS[tid][0];
      #pragma unroll
      for (int q = 1; q < 8; ++q) s = fmaxf(s, pmaxS[tid][q]);
      srowS[tid] = s;
    }
    __syncthreads();

    // ---- exp stage: expa = exp(alpha - s) as f16, stored K-contiguous ----
    {
      float s = srowS[m];
      v8h e8;
      e8[0] = (_Float16)__expf(a0.x - s); e8[1] = (_Float16)__expf(a0.y - s);
      e8[2] = (_Float16)__expf(a0.z - s); e8[3] = (_Float16)__expf(a0.w - s);
      e8[4] = (_Float16)__expf(a1.x - s); e8[5] = (_Float16)__expf(a1.y - s);
      e8[6] = (_Float16)__expf(a1.z - s); e8[7] = (_Float16)__expf(a1.w - s);
      *(v8h*)&expaS[m][j0] = e8;
    }
    __syncthreads();

    // ---- WMMA: D(16x16) = expA(16x64) x expT(64x16), f32 accumulate ----
    // 16-bit A 16x32 layout: lane L -> M = L&15; koff = (L>=16)?8:0;
    // element e -> K = (e>>3)*16 + koff + (e&7)  => two b128 LDS loads per half
    const int mrow = lane & 15;
    const int kof8 = (lane >> 4) * 8;
    v8h alo0 = *(const v8h*)&expaS[mrow][kof8];        // K = kof8..kof8+7
    v8h ahi0 = *(const v8h*)&expaS[mrow][16 + kof8];   // K = 16+kof8..
    v8h alo1 = *(const v8h*)&expaS[mrow][32 + kof8];
    v8h ahi1 = *(const v8h*)&expaS[mrow][48 + kof8];
    v16h af0, af1;
    #pragma unroll
    for (int e = 0; e < 8; ++e) {
      af0[e] = alo0[e]; af0[8 + e] = ahi0[e];
      af1[e] = alo1[e]; af1[8 + e] = ahi1[e];
    }
    v8f c = {};
    c = __builtin_amdgcn_wmma_f32_16x16x32_f16(false, af0, false, bf0,
                                               (short)0, c, false, false);
    c = __builtin_amdgcn_wmma_f32_16x16x32_f16(false, af1, false, bf1,
                                               (short)0, c, false, false);

    // ---- update: alpha' = mask*(s + log(D) + em) + (1-mask)*alpha ----
    // f32 D layout: element v -> M = v + (lane>=16?8:0), N = lane&15
    const int jw = wv * 16 + (lane & 15);
    const int mh = (lane >> 4) * 8;
    #pragma unroll
    for (int v = 0; v < 8; ++v) {
      int m2 = mh + v;
      float oldv = alphaS[m2][jw];
      float nw = srowS[m2] + __logf(fmaxf(c[v], 1e-30f)) + emS[t & 3][m2][jw];
      float mv = mvalS[m2];
      alphaS[m2][jw] = mv * nw + (1.0f - mv) * oldv;
    }
#if HAVE_ASYNC_LDS
    // Only this iteration's 2 async ops may stay in flight; the buffer read at
    // t+1 (issued at t-1) is then guaranteed complete. Tail: drain fully.
    if (t + 2 < SEQ) wait_async_le2(); else wait_async_le0();
#endif
    __syncthreads();
  }

  // ---- total[b] = logsumexp_i(alpha[i] + T[i][END]) ----
  if (tid < TILE_B) {
    float mx = -3.0e38f;
    for (int i = 0; i < NTAGS; ++i) {
      float x = alphaS[tid][i] + trans[i * NTAGS + END_ID];
      mx = fmaxf(mx, x);
    }
    float sum = 0.0f;
    for (int i = 0; i < NTAGS; ++i) {
      float x = alphaS[tid][i] + trans[i * NTAGS + END_ID];
      sum += __expf(x - mx);
    }
    totS[tid] = mx + __logf(sum);
  }
  __syncthreads();
  if (tid == 0) {
    float s = 0.0f;
    #pragma unroll
    for (int i = 0; i < TILE_B; ++i) s += totS[i];
    atomicAdd(out, s);
  }
}

// ---------------------------------------------------------------------------
// Gold (numerator) score: one thread per sequence, block-reduced, subtracted.
// ---------------------------------------------------------------------------
__global__ __launch_bounds__(256) void crf_gold(
    const float* __restrict__ emis,
    const int*   __restrict__ tags,
    const float* __restrict__ mask,
    const float* __restrict__ trans,
    float* __restrict__ out)
{
  const int b = blockIdx.x * blockDim.x + threadIdx.x;
  float g = 0.0f;
  if (b < NBATCH) {
    const int*   tg = tags + (size_t)b * SEQ;
    const float* mk = mask + (size_t)b * SEQ;
    const float* em = emis + (size_t)b * SEQ * NTAGS;
    int prev = tg[0];
    g = trans[START_ID * NTAGS + prev] + em[prev];
    float msum = mk[0];
    for (int t = 1; t < SEQ; ++t) {
      int   ct = tg[t];
      float mt = mk[t];
      g += (trans[prev * NTAGS + ct] + em[(size_t)t * NTAGS + ct]) * mt;
      msum += mt;
      prev = ct;
    }
    int lidx = (int)(msum - 1.0f);
    g += trans[tg[lidx] * NTAGS + END_ID];
  }
  __shared__ float red[256];
  red[threadIdx.x] = (b < NBATCH) ? g : 0.0f;
  __syncthreads();
  for (int st = 128; st > 0; st >>= 1) {
    if (threadIdx.x < st) red[threadIdx.x] += red[threadIdx.x + st];
    __syncthreads();
  }
  if (threadIdx.x == 0) atomicAdd(out, -red[0]);
}

extern "C" void kernel_launch(void* const* d_in, const int* in_sizes, int n_in,
                              void* d_out, int out_size, void* d_ws, size_t ws_size,
                              hipStream_t stream) {
  const float* emis  = (const float*)d_in[0];   // (512,512,64) f32
  const int*   tags  = (const int*)  d_in[1];   // (512,512) i32
  const float* mask  = (const float*)d_in[2];   // (512,512) f32
  const float* trans = (const float*)d_in[3];   // (64,64)   f32
  float* out = (float*)d_out;

  (void)hipMemsetAsync(out, 0, sizeof(float), stream);
  crf_gold<<<NBATCH / 256, 256, 0, stream>>>(emis, tags, mask, trans, out);
  crf_forward_wmma<<<NBATCH / TILE_B, 128, 0, stream>>>(emis, mask, trans, out);
}
